// GeoBiasAttention_33895881900037
// MI455X (gfx1250) — compile-verified, matrix operand order checked by reading, not run on hardware
//
#include <hip/hip_runtime.h>
#include <hip/hip_bf16.h>

// ---------------- problem constants ----------------
constexpr int DM  = 512;    // d_model
constexpr int NT  = 2048;   // sequence length
constexpr int NH  = 8;      // heads
constexpr int DH  = 64;     // head dim
constexpr int BH  = 32;     // bias hidden

typedef __attribute__((ext_vector_type(16))) _Float16     v16h;
typedef __attribute__((ext_vector_type(8)))  float        v8f;
typedef __attribute__((ext_vector_type(4)))  unsigned int u32x4;
typedef __attribute__((ext_vector_type(8)))  int          i32x8;
typedef __attribute__((ext_vector_type(4)))  int          i32x4;

// ---------------- WMMA helpers (CDNA5 ISA 7.12.2 layouts) ----------------
// A fragment: 16x32 f16. lane&15 = M row; lanes 16-31 use K offset +8.
// VGPR j holds K pair: K0 = 2*(j&3) + ((lane>>4)&1)*8 + (j&4)*4
__device__ __forceinline__ v16h load_frag_a(const _Float16* base, int ld, int lane) {
  const int m = lane & 15;
  const int ko = ((lane >> 4) & 1) * 8;
  v16h f;
#pragma unroll
  for (int j = 0; j < 8; ++j) {
    const int k = 2 * (j & 3) + ko + (j & 4) * 4;
    f[2 * j]     = base[m * ld + k];
    f[2 * j + 1] = base[m * ld + k + 1];
  }
  return f;
}

// B fragment: 32x16 f16, source stored "col-major" (one row per output col,
// K contiguous): element(K=k, col=n) = base[n*ld + k]
__device__ __forceinline__ v16h load_frag_b(const _Float16* base, int ld, int lane) {
  const int n = lane & 15;
  const int ko = ((lane >> 4) & 1) * 8;
  v16h f;
#pragma unroll
  for (int j = 0; j < 8; ++j) {
    const int k = 2 * (j & 3) + ko + (j & 4) * 4;
    f[2 * j]     = base[n * ld + k];
    f[2 * j + 1] = base[n * ld + k + 1];
  }
  return f;
}

__device__ __forceinline__ v8f wmma16(v16h a, v16h b, v8f c) {
  return __builtin_amdgcn_wmma_f32_16x16x32_f16(false, a, false, b, (short)0, c,
                                                false, false);
}

// ---------------- TDM: 2D f16 tile -> LDS (ISA 8.3/8.4 descriptor) --------
// Tensor dims are given relative to the tile-start address (== tile dims, no
// clipping); stride_d0 = row pitch in elements. lds_off = low 32 bits of the
// flat shared-memory address (aperture layout: addr[31:0] == LDS offset).
__device__ __forceinline__ void tdm_load_2d_f16(unsigned lds_off,
                                                const _Float16* gptr,
                                                unsigned tile_d0, unsigned tile_d1,
                                                unsigned stride_d0) {
  const unsigned long long ga = (unsigned long long)(size_t)(const void*)gptr;
  u32x4 g0;
  g0[0] = 1u;                                            // count=1, load, no gather
  g0[1] = lds_off;                                       // lds_addr (bytes)
  g0[2] = (unsigned)(ga & 0xffffffffull);                // global_addr[31:0]
  g0[3] = (unsigned)((ga >> 32) & 0x1ffffffull) | (2u << 30);  // addr[56:32], type=2
  i32x8 g1;
  g1[0] = (int)(1u << 16);                               // wg_mask=0, data_size=2B
  g1[1] = (int)((tile_d0 & 0xffffu) << 16);              // tensor_dim0[15:0]
  g1[2] = (int)((tile_d0 >> 16) | ((tile_d1 & 0xffffu) << 16)); // dim0 hi | dim1 lo
  g1[3] = (int)((tile_d1 >> 16) | (tile_d0 << 16));      // dim1 hi | tile_dim0
  g1[4] = (int)tile_d1;                                  // tile_dim1; tile_dim2=0
  g1[5] = (int)stride_d0;                                // tensor_dim0_stride[31:0]
  g1[6] = 0;                                             // stride hi; dim1_stride lo
  g1[7] = 0;
  i32x4 z4 = {0, 0, 0, 0};
#if defined(__clang_major__) && (__clang_major__ >= 23)
  i32x8 z8 = {0, 0, 0, 0, 0, 0, 0, 0};
  __builtin_amdgcn_tensor_load_to_lds(g0, g1, z4, z4, z8, 0);
#else
  __builtin_amdgcn_tensor_load_to_lds(g0, g1, z4, z4, 0);
#endif
}

__device__ __forceinline__ unsigned lds_off_of(const void* p) {
  return (unsigned)(size_t)p;   // low 32 bits of flat shared address = LDS offset
}

// ---------------- kernels ----------------
__global__ void k_cvt(const float* __restrict__ in, _Float16* __restrict__ out, int n) {
  int i = blockIdx.x * blockDim.x + threadIdx.x;
  if (i < n) out[i] = (_Float16)in[i];
}

// u[n][j] = dot(W1[j,:], xyz[n,:])   (layer-1 of bias MLP factored per token)
__global__ void k_u(const float* __restrict__ xyz, const float* __restrict__ W1,
                    float* __restrict__ u) {
  int i = blockIdx.x * blockDim.x + threadIdx.x;
  if (i >= NT * BH) return;
  int n = i >> 5, j = i & 31;
  const float* p = xyz + n * 3;
  const float* w = W1 + j * 3;
  u[i] = p[0] * w[0] + p[1] * w[1] + p[2] * w[2];
}

// One wave per 16x16 output tile: dst = xh @ Wh^T + b.  Both 16x512 operand
// strips are staged to LDS by the Tensor Data Mover, then 16 WMMAs run from
// LDS. transposed==1 stores dst[col][row] (v -> vT for K-contiguous AV B-frags)
__global__ __launch_bounds__(32) void k_qkv(const _Float16* __restrict__ xh,
                                            const _Float16* __restrict__ wh,
                                            const float* __restrict__ bias,
                                            _Float16* __restrict__ dst,
                                            int transposed) {
  __shared__ _Float16 At[16 * DM];   // 16 KB
  __shared__ _Float16 Bt[16 * DM];   // 16 KB
  const int nt = blockIdx.x;        // 0..127
  const int ct = blockIdx.y;        // 0..31
  const int lane = threadIdx.x;

  tdm_load_2d_f16(lds_off_of(At), xh + (size_t)nt * 16 * DM, DM, 16, DM);
  tdm_load_2d_f16(lds_off_of(Bt), wh + (size_t)ct * 16 * DM, DM, 16, DM);
  __builtin_amdgcn_s_wait_tensorcnt(0);

  v8f c = {};
#pragma unroll 4
  for (int kc = 0; kc < DM; kc += 32) {
    v16h a = load_frag_a(At + kc, DM, lane);
    v16h b = load_frag_b(Bt + kc, DM, lane);
    c = wmma16(a, b, c);
  }
  const int col0 = ct * 16 + (lane & 15);
  const float bv = bias[col0];
  const int rbase = nt * 16 + 8 * (lane >> 4);
  if (!transposed) {
#pragma unroll
    for (int r = 0; r < 8; ++r)
      dst[(rbase + r) * DM + col0] = (_Float16)(c[r] + bv);
  } else {
#pragma unroll
    for (int r = 0; r < 8; ++r)
      dst[col0 * NT + (rbase + r)] = (_Float16)(c[r] + bv);
  }
}

// Fused: scores (QK^T/8 + geo-bias MLP) -> softmax in LDS -> write attn ->
// AV GEMM from LDS. One block = (head, 16-query block), 4 waves.
// K tiles are TDM double-buffered per wave; score block lives in LDS.
__global__ __launch_bounds__(128) void k_attn(
    const _Float16* __restrict__ qh, const _Float16* __restrict__ kh,
    const _Float16* __restrict__ vT, const float* __restrict__ u,
    const float* __restrict__ W2, const float* __restrict__ b1,
    const float* __restrict__ b2, float* __restrict__ attn,
    _Float16* __restrict__ oh) {
  extern __shared__ char smem[];
  float*    S    = (float*)smem;                          // 16*2048 f32 (128 KB)
  _Float16* kbuf = (_Float16*)(smem + 16 * NT * 4);       // 4 waves x 2 x 16x64 f16
  float*    uqs  = (float*)((char*)kbuf + 4 * 2 * 16 * DH * 2);  // 16*32 f32
  float*    w2s  = uqs + 16 * BH;                         // 32
  float*    b1s  = w2s + BH;                              // 32
  float*    red  = b1s + BH;                              // 128
  float*    rinv = red + 128;                             // 16

  const int tid  = threadIdx.x;
  const int wave = tid >> 5;
  const int lane = tid & 31;
  const int h    = blockIdx.y;
  const int qb   = blockIdx.x;

  if (tid < BH) { w2s[tid] = W2[h * BH + tid]; b1s[tid] = b1[tid]; }
  for (int idx = tid; idx < 16 * BH; idx += 128)           // stage u[query rows]
    uqs[idx] = u[(qb * 16 + (idx >> 5)) * BH + (idx & 31)];
  const float b2h = b2[h];

  // Q fragments for this wave (16 rows x 64 K as two 16x32 frags)
  const _Float16* qbase = qh + (qb * 16) * DM + h * DH;
  const v16h qa0 = load_frag_a(qbase, DM, lane);
  const v16h qa1 = load_frag_a(qbase + 32, DM, lane);
  __syncthreads();

  const int mcol = lane & 15;          // key col within tile (fixed per lane)
  const int rsel = 8 * (lane >> 4);    // query-row group for C fragment

  // per-wave K-tile double buffer (wave-uniform scalars for the TDM descriptor)
  _Float16* kb[2] = { kbuf + (wave * 2 + 0) * 16 * DH,
                      kbuf + (wave * 2 + 1) * 16 * DH };
  const unsigned kb_off0 = (unsigned)__builtin_amdgcn_readfirstlane((int)lds_off_of(kb[0]));
  const unsigned kb_off1 = (unsigned)__builtin_amdgcn_readfirstlane((int)lds_off_of(kb[1]));
  const int hh = __builtin_amdgcn_readfirstlane(h);
  const int w4 = __builtin_amdgcn_readfirstlane(wave);

  // prime buffer 0 with this wave's first K tile
  tdm_load_2d_f16(kb_off0, kh + (size_t)(w4 * 16) * DM + hh * DH, DH, 16, DM);

  // ---- score pass: each wave takes every 4th key tile, TDM pipelined ----
  int buf = 0;
  for (int mt = wave; mt < NT / 16; mt += 4, buf ^= 1) {
    const int mtn = mt + 4;
    if (mtn < NT / 16) {
      tdm_load_2d_f16(buf ? kb_off0 : kb_off1,
                      kh + (size_t)(mtn * 16) * DM + hh * DH, DH, 16, DM);
      __builtin_amdgcn_s_wait_tensorcnt(1);   // current tile's DMA done
    } else {
      __builtin_amdgcn_s_wait_tensorcnt(0);
    }
    const _Float16* kt = kb[buf];
    v16h kb0 = load_frag_b(kt, DH, lane);
    v16h kb1 = load_frag_b(kt + 32, DH, lane);
    v8f c = {};
    c = wmma16(qa0, kb0, c);
    c = wmma16(qa1, kb1, c);

    float acc[8];
#pragma unroll
    for (int r = 0; r < 8; ++r) acc[r] = c[r] * 0.125f + b2h;   // 1/sqrt(64)

    const float* uk = u + (mt * 16 + mcol) * BH;
#pragma unroll 8
    for (int j = 0; j < BH; ++j) {
      const float ukj = uk[j], w2j = w2s[j], b1j = b1s[j];
#pragma unroll
      for (int r = 0; r < 8; ++r) {
        float t = uqs[(rsel + r) * BH + j] - ukj + b1j;
        t = t > 0.f ? t : 0.f;
        acc[r] += w2j * t;
      }
    }
#pragma unroll
    for (int r = 0; r < 8; ++r)
      S[(rsel + r) * NT + mt * 16 + mcol] = acc[r];
  }
  __syncthreads();

  // ---- softmax over the 16x2048 block in LDS (8 threads per row) ----
  const int CH   = NT / 8;             // 256 elements per thread
  const int row  = tid >> 3;
  const int part = tid & 7;
  float* Sr = S + row * NT + part * CH;

  float mx = -3.0e38f;
  for (int i = 0; i < CH; ++i) mx = fmaxf(mx, Sr[i]);
  red[tid] = mx;
  __syncthreads();
  float rm = red[row * 8];
  for (int t = 1; t < 8; ++t) rm = fmaxf(rm, red[row * 8 + t]);
  __syncthreads();

  float sm = 0.f;
  for (int i = 0; i < CH; ++i) { float e = __expf(Sr[i] - rm); Sr[i] = e; sm += e; }
  red[tid] = sm;
  __syncthreads();
  sm = 0.f;
  for (int t = 0; t < 8; ++t) sm += red[row * 8 + t];
  if (part == 0) rinv[row] = 1.0f / sm;
  __syncthreads();

  // normalize in place + single coalesced write of attn[h][qb*16+r][m]
  float* abase = attn + ((size_t)h * NT + qb * 16) * NT;
  for (int idx = tid; idx < 16 * NT; idx += 128) {
    const int rr = idx >> 11, cc = idx & (NT - 1);
    const float p = S[rr * NT + cc] * rinv[rr];
    S[rr * NT + cc] = p;
    abase[(size_t)rr * NT + cc] = p;
  }
  __syncthreads();

  // ---- AV: each wave owns a 16-col group of Dh; 64 WMMAs over K=2048 ----
  const int cg = wave;
  v8f o = {};
  const _Float16* vbase = vT + (size_t)(h * DH + cg * 16) * NT;
  const int ko = ((lane >> 4) & 1) * 8;
  for (int kc = 0; kc < NT; kc += 32) {
    if (kc + 32 < NT)
      __builtin_prefetch(vbase + (lane & 15) * NT + kc + 32, 0, 1);
    v16h a;
#pragma unroll
    for (int j = 0; j < 8; ++j) {       // A-frag from LDS, f32 -> f16
      const int k = kc + 2 * (j & 3) + ko + (j & 4) * 4;
      a[2 * j]     = (_Float16)S[mcol * NT + k];
      a[2 * j + 1] = (_Float16)S[mcol * NT + k + 1];
    }
    v16h b = load_frag_b(vbase + kc, NT, lane);
    o = wmma16(a, b, o);
  }
  const int col0 = h * DH + cg * 16 + mcol;
  const int rbase = qb * 16 + rsel;
#pragma unroll
  for (int r = 0; r < 8; ++r)
    oh[(rbase + r) * DM + col0] = (_Float16)o[r];
}

// out = oh @ Wo^T + bo  (f32 result), TDM-staged operands like k_qkv
__global__ __launch_bounds__(32) void k_out(const _Float16* __restrict__ ohh,
                                            const _Float16* __restrict__ woh,
                                            const float* __restrict__ bo,
                                            float* __restrict__ out) {
  __shared__ _Float16 At[16 * DM];
  __shared__ _Float16 Bt[16 * DM];
  const int nt = blockIdx.x, ct = blockIdx.y, lane = threadIdx.x;

  tdm_load_2d_f16(lds_off_of(At), ohh + (size_t)nt * 16 * DM, DM, 16, DM);
  tdm_load_2d_f16(lds_off_of(Bt), woh + (size_t)ct * 16 * DM, DM, 16, DM);
  __builtin_amdgcn_s_wait_tensorcnt(0);

  v8f c = {};
#pragma unroll 4
  for (int kc = 0; kc < DM; kc += 32) {
    v16h a = load_frag_a(At + kc, DM, lane);
    v16h b = load_frag_b(Bt + kc, DM, lane);
    c = wmma16(a, b, c);
  }
  const int col0 = ct * 16 + (lane & 15);
  const float bv = bo[col0];
  const int rbase = nt * 16 + 8 * (lane >> 4);
#pragma unroll
  for (int r = 0; r < 8; ++r)
    out[(rbase + r) * DM + col0] = c[r] + bv;
}

// ---------------- host launcher ----------------
extern "C" void kernel_launch(void* const* d_in, const int* in_sizes, int n_in,
                              void* d_out, int out_size, void* d_ws, size_t ws_size,
                              hipStream_t stream) {
  const float* x   = (const float*)d_in[0];
  const float* xyz = (const float*)d_in[1];
  const float* Wq  = (const float*)d_in[2];
  const float* bq  = (const float*)d_in[3];
  const float* Wk  = (const float*)d_in[4];
  const float* bk  = (const float*)d_in[5];
  const float* Wv  = (const float*)d_in[6];
  const float* bv  = (const float*)d_in[7];
  const float* Wo  = (const float*)d_in[8];
  const float* bo  = (const float*)d_in[9];
  const float* W1  = (const float*)d_in[10];
  const float* b1  = (const float*)d_in[11];
  const float* W2  = (const float*)d_in[12];
  const float* b2  = (const float*)d_in[13];

  char* ws = (char*)d_ws;
  const size_t MB = 1024 * 1024;
  _Float16* xh  = (_Float16*)(ws + 0 * MB);          // 2 MB : x f16
  _Float16* wqh = (_Float16*)(ws + 2 * MB);
  _Float16* wkh = (_Float16*)(ws + 2 * MB + 512 * 1024);
  _Float16* wvh = (_Float16*)(ws + 3 * MB);
  _Float16* woh = (_Float16*)(ws + 3 * MB + 512 * 1024);
  _Float16* qh  = (_Float16*)(ws + 4 * MB);          // 2 MB
  _Float16* kh  = (_Float16*)(ws + 6 * MB);          // 2 MB
  _Float16* vT  = (_Float16*)(ws + 8 * MB);          // 2 MB (transposed [d][m])
  _Float16* oh  = (_Float16*)(ws + 10 * MB);         // 2 MB
  float*    u   = (float*)(ws + 12 * MB);            // 256 KB

  float* out_p  = (float*)d_out;                     // [NT][DM]
  float* attn_p = out_p + (size_t)NT * DM;           // [NH][NT][NT]

  // 1) converts + per-token bias projection
  k_cvt<<<dim3((NT * DM + 255) / 256), dim3(256), 0, stream>>>(x, xh, NT * DM);
  k_cvt<<<dim3((DM * DM + 255) / 256), dim3(256), 0, stream>>>(Wq, wqh, DM * DM);
  k_cvt<<<dim3((DM * DM + 255) / 256), dim3(256), 0, stream>>>(Wk, wkh, DM * DM);
  k_cvt<<<dim3((DM * DM + 255) / 256), dim3(256), 0, stream>>>(Wv, wvh, DM * DM);
  k_cvt<<<dim3((DM * DM + 255) / 256), dim3(256), 0, stream>>>(Wo, woh, DM * DM);
  k_u<<<dim3((NT * BH + 255) / 256), dim3(256), 0, stream>>>(xyz, W1, u);

  // 2) projections (one wave per 16x16 tile, TDM-staged operands)
  dim3 gproj(NT / 16, DM / 16);
  k_qkv<<<gproj, dim3(32), 0, stream>>>(xh, wqh, bq, qh, 0);
  k_qkv<<<gproj, dim3(32), 0, stream>>>(xh, wkh, bk, kh, 0);
  k_qkv<<<gproj, dim3(32), 0, stream>>>(xh, wvh, bv, vT, 1);

  // 3) fused bias+scores+softmax+AV  (~147 KB dynamic LDS)
  const size_t smem = (size_t)16 * NT * 4            // S
                    + (size_t)4 * 2 * 16 * DH * 2    // kbuf
                    + (size_t)16 * BH * 4            // uqs
                    + (BH + BH + 128 + 16) * 4;      // w2s/b1s/red/rinv
  k_attn<<<dim3(NT / 16, NH), dim3(128), smem, stream>>>(
      qh, kh, vT, u, W2, b1, b2, attn_p, oh);

  // 4) output projection
  k_out<<<gproj, dim3(32), 0, stream>>>(oh, woh, bo, out_p);
}
